// AxialBlock_10136122819265
// MI455X (gfx1250) — compile-verified, matrix operand order checked by reading, not run on hardware
//
#include <hip/hip_runtime.h>
#include <hip/hip_bf16.h>

typedef __attribute__((ext_vector_type(16))) __bf16 v16bf;
typedef __attribute__((ext_vector_type(8)))  __bf16 v8bf;
typedef __attribute__((ext_vector_type(4)))  __bf16 v4bf;
typedef __attribute__((ext_vector_type(8)))  float  v8f;

// ---------------------------------------------------------------------------
// helpers
// ---------------------------------------------------------------------------
static __device__ inline v16bf pack16(v8bf lo, v8bf hi) {
    v16bf r;
#pragma unroll
    for (int e = 0; e < 8; ++e) { r[e] = lo[e]; r[e + 8] = hi[e]; }
    return r;
}

// Load a 16-element bf16 fragment for one lane: two contiguous 8-half chunks
// at +0 and +16 halves (matches CDNA5 16-bit A/B VGPR layout where the lane's
// K-chunks are [8g..8g+7] and [16+8g..16+8g+7]; caller adds the 8*g offset).
static __device__ inline v16bf load_frag(const __bf16* p) {
    v8bf lo = *(const v8bf*)p;
    v8bf hi = *(const v8bf*)(p + 16);
    return pack16(lo, hi);
}

static __device__ inline v8f wmma_bf16(v16bf a, v16bf b, v8f c) {
    return __builtin_amdgcn_wmma_f32_16x16x32_bf16(
        /*neg_a=*/false, a, /*neg_b=*/false, b,
        /*c_mod=*/(short)0, c, /*reuse_a=*/false, /*reuse_b=*/false);
}

// ---------------------------------------------------------------------------
// pack x: (B=4, C=512, S=16384) f32 -> token-major bf16 X[n=b*S+s][c]
// ---------------------------------------------------------------------------
__global__ __launch_bounds__(256) void pack_x_kernel(const float* __restrict__ x,
                                                     __bf16* __restrict__ xt) {
    __shared__ float tile[32][33];
    const int b = blockIdx.z, c0 = blockIdx.y * 32, s0 = blockIdx.x * 32;
    const int tx = threadIdx.x, ty = threadIdx.y;  // 32 x 8
#pragma unroll
    for (int j = 0; j < 4; ++j) {
        int cc = ty * 4 + j;
        tile[cc][tx] = x[(size_t)(b * 512 + c0 + cc) * 16384 + s0 + tx];
    }
    __syncthreads();
#pragma unroll
    for (int j = 0; j < 4; ++j) {
        int sr = ty * 4 + j;
        xt[(size_t)(b * 16384 + s0 + sr) * 512 + c0 + tx] = (__bf16)tile[tx][sr];
    }
}

// ---------------------------------------------------------------------------
// weight convert+transpose: w f32 [cin][cout] -> wt bf16 [cout][cin]
// ---------------------------------------------------------------------------
__global__ __launch_bounds__(256) void conv_wt_kernel(const float* __restrict__ w,
                                                      __bf16* __restrict__ wt) {
    int id = blockIdx.x * 256 + threadIdx.x;   // 512*512 threads
    int co = id >> 9, ci = id & 511;
    wt[((size_t)co << 9) | ci] = (__bf16)w[((size_t)ci << 9) + co];
}

// ---------------------------------------------------------------------------
// GEMM: out[M=grid.x*64][512] = A[M][512] * Wt^T  (Wt is [N=512][K=512])
// mode 0: outB = bf16(acc*scale)
// mode 1: outF = acc + bias
// mode 2: outF += acc + bias
// ---------------------------------------------------------------------------
__global__ __launch_bounds__(256) void gemm512_kernel(
    const __bf16* __restrict__ A, const __bf16* __restrict__ Bt,
    const float* __restrict__ bias, __bf16* __restrict__ outB,
    float* __restrict__ outF, int mode, float scale) {
    const int wave = threadIdx.x >> 5, lane = threadIdx.x & 31;
    const int l16 = lane & 15, g = lane >> 4;
    const int mi = wave & 1, ni = wave >> 1;            // 2x4 wave grid
    const int mB = blockIdx.x * 64 + mi * 32;
    const int nB = blockIdx.y * 256 + ni * 64;

    v8f acc[2][4];
#pragma unroll
    for (int i = 0; i < 2; ++i)
#pragma unroll
        for (int j = 0; j < 4; ++j)
#pragma unroll
            for (int r = 0; r < 8; ++r) acc[i][j][r] = 0.0f;

    for (int k0 = 0; k0 < 512; k0 += 32) {
        v16bf a[2], b[4];
#pragma unroll
        for (int t = 0; t < 2; ++t)
            a[t] = load_frag(A + (size_t)(mB + t * 16 + l16) * 512 + k0 + 8 * g);
#pragma unroll
        for (int t = 0; t < 4; ++t)
            b[t] = load_frag(Bt + (size_t)(nB + t * 16 + l16) * 512 + k0 + 8 * g);
#pragma unroll
        for (int i = 0; i < 2; ++i)
#pragma unroll
            for (int j = 0; j < 4; ++j)
                acc[i][j] = wmma_bf16(a[i], b[j], acc[i][j]);
    }

#pragma unroll
    for (int i = 0; i < 2; ++i)
#pragma unroll
        for (int j = 0; j < 4; ++j)
#pragma unroll
            for (int r = 0; r < 8; ++r) {
                // C/D layout: VGPR r -> M=r (lanes 0-15) / M=8+r (lanes 16-31)
                int row = mB + i * 16 + g * 8 + r;
                int col = nB + j * 16 + l16;
                float v = acc[i][j][r];
                if (mode == 0) {
                    outB[(size_t)row * 512 + col] = (__bf16)(v * scale);
                } else {
                    float t = v + bias[col];
                    if (mode == 2) t += outF[(size_t)row * 512 + col];
                    outF[(size_t)row * 512 + col] = t;
                }
            }
}

// ---------------------------------------------------------------------------
// axial attention: one wave per (sequence, head). Q already scaled by 1/sqrt(d).
// axis 0: W (L=32, stride=1)  axis 1: H (L=32, stride=32)  axis 2: T (L=16, stride=1024)
// ---------------------------------------------------------------------------
__global__ __launch_bounds__(256) void axial_attn_kernel(
    const __bf16* __restrict__ Q, const __bf16* __restrict__ K,
    const __bf16* __restrict__ V, __bf16* __restrict__ O, int axis) {
    // per-wave LDS: P (32 rows x 40 pitch) + Vt (64 rows x 40 pitch), bf16
    __shared__ __bf16 smem[8 * 3840];
    const int wave = threadIdx.x >> 5, lane = threadIdx.x & 31;
    const int l16 = lane & 15, g = lane >> 4;
    const int unit = blockIdx.x * 8 + wave;
    const int head = unit & 7, seq = unit >> 3;

    int L, stride, seqbase;
    if (axis == 0)      { L = 32; stride = 1;    seqbase = seq * 32; }
    else if (axis == 1) { L = 32; stride = 32;   seqbase = (seq >> 5) * 1024 + (seq & 31); }
    else                { L = 16; stride = 1024; seqbase = (seq >> 10) * 16384 + (seq & 1023); }
    const int LT = L >> 4;
    const int hoff = head * 64;

    __bf16* P  = smem + wave * 3840;
    __bf16* Vt = P + 1280;

    // stage V transposed into LDS: Vt[channel][token]
    for (int idx = lane; idx < L * 16; idx += 32) {
        int tok = idx >> 4, cq = (idx & 15) * 4;
        const __bf16* vp = V + (size_t)(seqbase + tok * stride) * 512 + hoff + cq;
        v4bf vv = *(const v4bf*)vp;
#pragma unroll
        for (int e = 0; e < 4; ++e) Vt[(cq + e) * 40 + tok] = vv[e];
    }
    if (L == 16) {  // zero K-dim padding tokens 16..31 so padded WMMA stays exact
        for (int idx = lane; idx < 64 * 16; idx += 32) {
            int ch = idx >> 4, tok = 16 + (idx & 15);
            Vt[ch * 40 + tok] = (__bf16)0.0f;
        }
    }

    // S = Q K^T  (M=N=L, K=64)
    v8f s[2][2];
#pragma unroll
    for (int i = 0; i < 2; ++i)
#pragma unroll
        for (int j = 0; j < 2; ++j)
#pragma unroll
            for (int r = 0; r < 8; ++r) s[i][j][r] = 0.0f;

    for (int kk = 0; kk < 64; kk += 32) {
        v16bf qa[2], kb[2];
        for (int t = 0; t < LT; ++t) {
            qa[t] = load_frag(Q + (size_t)(seqbase + (t * 16 + l16) * stride) * 512 + hoff + kk + 8 * g);
            kb[t] = load_frag(K + (size_t)(seqbase + (t * 16 + l16) * stride) * 512 + hoff + kk + 8 * g);
        }
        for (int i = 0; i < LT; ++i)
            for (int j = 0; j < LT; ++j)
                s[i][j] = wmma_bf16(qa[i], kb[j], s[i][j]);
    }

    // row softmax (rows live in 16-lane halves of the wave) -> P in LDS (A-layout source)
    for (int i = 0; i < LT; ++i) {
#pragma unroll
        for (int r = 0; r < 8; ++r) {
            float v0 = s[i][0][r];
            float v1 = (LT == 2) ? s[i][1][r] : -3.4e38f;
            float m = fmaxf(v0, v1);
#pragma unroll
            for (int msk = 1; msk < 16; msk <<= 1) m = fmaxf(m, __shfl_xor(m, msk, 32));
            float e0 = __expf(v0 - m);
            float e1 = (LT == 2) ? __expf(v1 - m) : 0.0f;
            float sm = e0 + e1;
#pragma unroll
            for (int msk = 1; msk < 16; msk <<= 1) sm += __shfl_xor(sm, msk, 32);
            float inv = 1.0f / sm;
            int row = i * 16 + g * 8 + r;
            P[row * 40 + l16] = (__bf16)(e0 * inv);
            if (LT == 2) P[row * 40 + 16 + l16] = (__bf16)(e1 * inv);
            else         P[row * 40 + 16 + l16] = (__bf16)0.0f;  // zero pad K-dim
        }
    }
    __syncthreads();

    // O = P @ V  (M=L, K=32 padded, N=64)
    v16bf pa[2], vb[4];
    for (int t = 0; t < LT; ++t) {
        int row = t * 16 + l16;
        pa[t] = load_frag(P + row * 40 + 8 * g);
    }
#pragma unroll
    for (int j = 0; j < 4; ++j) {
        int ch = j * 16 + l16;
        vb[j] = load_frag(Vt + ch * 40 + 8 * g);
    }
    for (int i = 0; i < LT; ++i) {
#pragma unroll
        for (int j = 0; j < 4; ++j) {
            v8f o;
#pragma unroll
            for (int r = 0; r < 8; ++r) o[r] = 0.0f;
            o = wmma_bf16(pa[i], vb[j], o);
#pragma unroll
            for (int r = 0; r < 8; ++r) {
                int tokrow = i * 16 + g * 8 + r;
                size_t tok = (size_t)(seqbase + tokrow * stride);
                O[tok * 512 + hoff + j * 16 + l16] = (__bf16)o[r];
            }
        }
    }
}

// ---------------------------------------------------------------------------
// unpack: accum f32 [n][c] -> out f32 (B,C,T,H,W)
// ---------------------------------------------------------------------------
__global__ __launch_bounds__(256) void unpack_out_kernel(const float* __restrict__ acc,
                                                         float* __restrict__ out) {
    __shared__ float tile[32][33];
    const int b = blockIdx.z, c0 = blockIdx.y * 32, s0 = blockIdx.x * 32;
    const int tx = threadIdx.x, ty = threadIdx.y;
#pragma unroll
    for (int j = 0; j < 4; ++j) {
        int sr = ty * 4 + j;
        tile[tx][sr] = acc[(size_t)(b * 16384 + s0 + sr) * 512 + c0 + tx];
    }
    __syncthreads();
#pragma unroll
    for (int j = 0; j < 4; ++j) {
        int cc = ty * 4 + j;
        out[(size_t)(b * 512 + c0 + cc) * 16384 + s0 + tx] = tile[cc][tx];
    }
}

// ---------------------------------------------------------------------------
// launch
// ---------------------------------------------------------------------------
extern "C" void kernel_launch(void* const* d_in, const int* in_sizes, int n_in,
                              void* d_out, int out_size, void* d_ws, size_t ws_size,
                              hipStream_t stream) {
    (void)in_sizes; (void)n_in; (void)out_size; (void)ws_size;
    const float* x = (const float*)d_in[0];

    constexpr size_t MB64 = 67108864ull;          // 65536*512*2 bytes
    char* ws = (char*)d_ws;
    __bf16* xt    = (__bf16*)(ws + 0 * MB64);
    __bf16* qbuf  = (__bf16*)(ws + 1 * MB64);
    __bf16* kbuf  = (__bf16*)(ws + 2 * MB64);
    __bf16* vbuf  = (__bf16*)(ws + 3 * MB64);
    __bf16* obuf  = (__bf16*)(ws + 4 * MB64);
    float*  accum = (float*) (ws + 5 * MB64);     // 65536*512*4 = 2*MB64
    __bf16* wtq   = (__bf16*)(ws + 7 * MB64);
    __bf16* wtk   = wtq + 512 * 512;
    __bf16* wtv   = wtk + 512 * 512;
    __bf16* wtf   = wtv + 512 * 512;

    // pack x -> token-major bf16
    pack_x_kernel<<<dim3(512, 16, 4), dim3(32, 8), 0, stream>>>(x, xt);

    const dim3 ggrid(1024, 2), gblk(256);
    for (int br = 0; br < 3; ++br) {            // 0=W, 1=H, 2=T (input order w,h,t)
        int base = 1 + br * 5;
        conv_wt_kernel<<<1024, 256, 0, stream>>>((const float*)d_in[base + 0], wtq);
        conv_wt_kernel<<<1024, 256, 0, stream>>>((const float*)d_in[base + 1], wtk);
        conv_wt_kernel<<<1024, 256, 0, stream>>>((const float*)d_in[base + 2], wtv);
        conv_wt_kernel<<<1024, 256, 0, stream>>>((const float*)d_in[base + 3], wtf);
        const float* fb = (const float*)d_in[base + 4];

        // QKV projections (Q carries the 1/sqrt(64) scale)
        gemm512_kernel<<<ggrid, gblk, 0, stream>>>(xt, wtq, nullptr, qbuf, nullptr, 0, 0.125f);
        gemm512_kernel<<<ggrid, gblk, 0, stream>>>(xt, wtk, nullptr, kbuf, nullptr, 0, 1.0f);
        gemm512_kernel<<<ggrid, gblk, 0, stream>>>(xt, wtv, nullptr, vbuf, nullptr, 0, 1.0f);

        // axial attention: one wave per (seq, head); 8 waves per block
        int blocks = (br == 2) ? 4096 : 2048;
        axial_attn_kernel<<<blocks, 256, 0, stream>>>(qbuf, kbuf, vbuf, obuf, br);

        // output projection + bias, accumulate branches into f32
        gemm512_kernel<<<ggrid, gblk, 0, stream>>>(obuf, wtf, fb, nullptr, accum,
                                                   (br == 0) ? 1 : 2, 1.0f);
    }

    // transpose back to (B, C, T, H, W)
    unpack_out_kernel<<<dim3(512, 16, 4), dim3(32, 8), 0, stream>>>(accum, (float*)d_out);
}
